// ISTFT_77738908058000
// MI455X (gfx1250) — matrix operand order repaired and checked by reading, not compiled, not run on hardware
//
#include <hip/hip_runtime.h>
#include <hip/hip_bf16.h>

typedef __attribute__((ext_vector_type(16))) __bf16 v16bf;
typedef __attribute__((ext_vector_type(8)))  float  v8f;

// Problem constants
#define N_FFT   1024
#define FBINS   513
#define TFRM    4096
#define HOP     256
#define PADC    384
#define OUTL    1048576        // per-batch cropped output length
#define KPAD    1056           // 33 * 32
#define KTILES  33
#define NTILES  64             // 1024 / 16
#define BPLANE  (KTILES * NTILES * 512)   // ushorts per basis plane (hi or lo)

// ---- bf16 split helpers (bit-exact, no __bf16 arithmetic needed) ----
__device__ __forceinline__ unsigned short f2bf(float f) {
    unsigned u = __float_as_uint(f);
    unsigned r = u + 0x7FFFu + ((u >> 16) & 1u);   // round-to-nearest-even
    return (unsigned short)(r >> 16);
}
__device__ __forceinline__ float bf2f(unsigned short h) {
    return __uint_as_float(((unsigned)h) << 16);
}

// =====================================================================
// Kernel 1: build windowed DFT basis, split into bf16 hi/lo planes,
// pre-swizzled for V_WMMA_F32_16X16X32_BF16 B in this file's contiguous k order:
//   B frag (32x16): lanes 0-15 hold K=0..15 (elem e == K), lanes 16-31
//   hold K=16..31.  Stored so each lane reads 16 contiguous bf16 (32B).
// ws layout: [hi plane: BPLANE ushort][lo plane: BPLANE ushort]
// =====================================================================
__global__ __launch_bounds__(256) void istft_prep_basis(
    const float* __restrict__ win, unsigned short* __restrict__ basis)
{
    int idx = blockIdx.x * 256 + threadIdx.x;
    if (idx >= BPLANE) return;
    int e    = idx & 15;
    int lane = (idx >> 4) & 31;
    int rest = idx >> 9;
    int nt   = rest % NTILES;
    int kt   = rest / NTILES;

    int k = kt * 32 + ((lane & 16) ? 16 : 0) + e;   // global K row
    int n = nt * 16 + (lane & 15);                  // global N column

    float v = 0.0f;
    if (k < FBINS) {
        // real-part rows: a_k * cos(2*pi*k*n/N)
        float a  = (k == 0 || k == 512) ? (1.0f / 1024.0f) : (2.0f / 1024.0f);
        int   ph = (k * n) & (N_FFT - 1);           // exact phase reduction
        v = a * cosf((float)ph * 0.006135923151542565f);
    } else if (k < 2 * FBINS) {
        // imag-part rows: -2/N * sin(2*pi*kk*n/N); bins 0 and 512 ignored
        int kk = k - FBINS;
        if (kk != 0 && kk != 512) {
            int ph = (kk * n) & (N_FFT - 1);
            v = -(2.0f / 1024.0f) * sinf((float)ph * 0.006135923151542565f);
        }
    } // k >= 1026: zero padding rows

    v *= win[n];                                    // fold window into basis

    unsigned short hi = f2bf(v);
    unsigned short lo = f2bf(v - bf2f(hi));
    basis[idx]          = hi;
    basis[BPLANE + idx] = lo;
}

// =====================================================================
// Kernel 2: fused GEMM (irfft*window) + overlap-add + envelope divide.
// Block = (batch b, 16 hops h0..h0+15) -> 4096 output samples.
// 8 waves: wave w -> chunk j = w>>1 (frame t = h - j, cols [256j,256j+256)),
//          column half = w&1.  Per wave: 8 16x16 f32 accum tiles,
//          K-loop 33 x {1 A-frag pair from LDS, 8 B-frag pairs from L2,
//          24 v_wmma_f32_16x16x32_bf16}.
// =====================================================================
__global__ __launch_bounds__(256) void istft_main(
    const float* __restrict__ spec_re,
    const float* __restrict__ spec_im,
    const float* __restrict__ win,
    const unsigned short* __restrict__ basis,
    float* __restrict__ out)
{
    // A staging: 19 frame-rows (t = h0-3 .. h0+15) x 32 k, hi/lo bf16,
    // k in natural order, so each lane's fragment is 16 contiguous bf16.
    __shared__ __align__(32) unsigned short sA[2][19 * 32];
    __shared__ float sOut[16 * 256];                // cross-chunk OLA tile

    const int b    = blockIdx.y;
    const int h0   = blockIdx.x * 16;
    const int tid  = threadIdx.x;
    const int lane = tid & 31;
    const int w    = tid >> 5;
    const int j    = w >> 1;        // chunk 0..3
    const int ch   = w & 1;         // column half 0..1

    for (int i = tid; i < 16 * 256; i += 256) sOut[i] = 0.0f;

    v8f acc[8] = {};

    const int m    = lane & 15;             // tile row
    const int r    = m - j + 3;             // staging row (0..18)
    const int koff = (lane & 16) ? 16 : 0;  // A/B fragment lane half

    for (int kt = 0; kt < KTILES; ++kt) {
        __syncthreads();
        // ---- cooperative A staging: f32 -> (hi,lo) bf16, k in natural order ----
        for (int idx = tid; idx < 19 * 32; idx += 256) {
            int kk = idx / 19;              // 0..31 within K-step
            int rr = idx % 19;              // staging row
            int t  = h0 - 3 + rr;           // frame index
            int kg = kt * 32 + kk;          // global K
            float v = 0.0f;
            if ((unsigned)t < (unsigned)TFRM) {
                if (kg < FBINS)
                    v = spec_re[((size_t)b * FBINS + kg) * TFRM + t];
                else if (kg < 2 * FBINS)
                    v = spec_im[((size_t)b * FBINS + (kg - FBINS)) * TFRM + t];
            }
            unsigned short hi = f2bf(v);
            unsigned short lo = f2bf(v - bf2f(hi));
            // A 16x32 bf16 layout: lane<16 holds K 0..15,
            // lane>=16 holds K 16..31 (contiguous, matches B).
            int kp = kk;
            sA[0][rr * 32 + kp] = hi;
            sA[1][rr * 32 + kp] = lo;
        }
        __syncthreads();

        // ---- A fragments: one 32B contiguous LDS read per plane ----
        const v16bf a_hi = *(const v16bf*)&sA[0][r * 32 + koff];
        const v16bf a_lo = *(const v16bf*)&sA[1][r * 32 + koff];

        // ---- 8 N-tiles: B frags from L2-resident swizzled basis ----
#pragma unroll
        for (int it = 0; it < 8; ++it) {
            int nt = j * 16 + ch * 8 + it;                  // global N-tile
            size_t off = ((size_t)(kt * NTILES + nt)) * 512 + (size_t)lane * 16;
            v16bf b_hi = *(const v16bf*)&basis[off];
            v16bf b_lo = *(const v16bf*)&basis[BPLANE + off];
            // f32-via-bf16 split: hi*hi + hi*lo + lo*hi
            acc[it] = __builtin_amdgcn_wmma_f32_16x16x32_bf16(
                false, a_hi, false, b_hi, (short)0, acc[it], false, false);
            acc[it] = __builtin_amdgcn_wmma_f32_16x16x32_bf16(
                false, a_hi, false, b_lo, (short)0, acc[it], false, false);
            acc[it] = __builtin_amdgcn_wmma_f32_16x16x32_bf16(
                false, a_lo, false, b_hi, (short)0, acc[it], false, false);
        }
    }

    __syncthreads();
    // ---- cross-chunk overlap-add reduction into LDS ----
    {
        const int mb = (lane & 16) ? 8 : 0;     // C/D layout: M = r + (lane<16?0:8)
#pragma unroll
        for (int it = 0; it < 8; ++it) {
            int ncol = (ch * 8 + it) * 16 + (lane & 15);   // 0..255 within hop
#pragma unroll
            for (int rr = 0; rr < 8; ++rr) {
                atomicAdd(&sOut[(mb + rr) * 256 + ncol], acc[it][rr]);
            }
        }
    }
    __syncthreads();

    // ---- envelope divide, crop, store (disjoint ranges per block) ----
    for (int li = tid; li < 16 * 256; li += 256) {
        int mm = li >> 8;
        int nc = li & 255;
        int h  = h0 + mm;
        float env = 0.0f;
#pragma unroll
        for (int jj = 0; jj < 4; ++jj) {
            int t = h - jj;
            if (t >= 0 && t < TFRM) {
                float wv = win[nc + jj * HOP];
                env += wv * wv;
            }
        }
        long s = (long)h * HOP + nc - PADC;     // cropped sample index
        if (s >= 0 && s < (long)OUTL) {
            out[(size_t)b * OUTL + (size_t)s] = sOut[li] / fmaxf(env, 1e-11f);
        }
    }
}

extern "C" void kernel_launch(void* const* d_in, const int* in_sizes, int n_in,
                              void* d_out, int out_size, void* d_ws, size_t ws_size,
                              hipStream_t stream) {
    const float* spec_re = (const float*)d_in[0];
    const float* spec_im = (const float*)d_in[1];
    const float* win     = (const float*)d_in[2];
    unsigned short* basis = (unsigned short*)d_ws;   // 2 * BPLANE ushorts (~4.3 MB)
    float* out = (float*)d_out;

    // Build windowed split-bf16 DFT basis every call (deterministic).
    int prep_blocks = (BPLANE + 255) / 256;
    istft_prep_basis<<<prep_blocks, 256, 0, stream>>>(win, basis);

    // Fused irfft-GEMM + OLA.  h ranges 0..4111 covers all cropped samples.
    dim3 grid(257, 8);
    istft_main<<<grid, 256, 0, stream>>>(spec_re, spec_im, win, basis, out);
}
